// EGNNLayer_26465588478020
// MI455X (gfx1250) — compile-verified, matrix operand order checked by reading, not run on hardware
//
#include <hip/hip_runtime.h>

// ---------------- problem constants ----------------
#define N_NODES 50000
#define N_EDGES 800000
#define D 128
#define EPS 1e-5f

// padded LDS strides (avoid 64-bank conflicts on 512B/256B row strides)
#define LDK  264   // K-stride (halves) for K=256 tiles  (528B, 16B aligned)
#define LDK2 136   // K-stride (halves) for K=128 tiles  (272B)
#define LDY  132   // row stride (floats) for layernorm staging

#define WAVES 8    // 256 threads / block = 8 wave32

typedef __attribute__((ext_vector_type(16))) _Float16 v16h;
typedef __attribute__((ext_vector_type(8)))  _Float16 v8h;
typedef __attribute__((ext_vector_type(4)))  _Float16 v4h;
typedef __attribute__((ext_vector_type(8)))  float    v8f;
typedef __attribute__((ext_vector_type(4)))  unsigned int v4u;
typedef __attribute__((ext_vector_type(8)))  unsigned int v8u;

__device__ __forceinline__ void lds_fence() {
  asm volatile("s_wait_dscnt 0" ::: "memory");
}
__device__ __forceinline__ void async_fence() {
  asm volatile("s_wait_asynccnt 0" ::: "memory");
}

// ---- Tensor Data Mover: 2D tile (global -> LDS) with LDS row padding ----
// D# packing per CDNA5 ISA 8.3/8.4 (2D tensor -> 2 SGPR groups).
// dims/strides in data_size units (data_size = 2 bytes, f16).
// pad_interval code: pad after 2^(code+1) dwords; pad_amount code: (code+1) dwords.
__device__ __forceinline__ void tdm_load_2d(unsigned lds_off, const void* gptr,
                                            unsigned dim0, unsigned dim1,
                                            unsigned stride0,
                                            unsigned pad_interval,
                                            unsigned pad_amount) {
  unsigned long long ga = (unsigned long long)gptr;
  v4u g0;
  g0[0] = 1u;                                   // count=1, load descriptor
  g0[1] = lds_off;                              // LDS byte address
  g0[2] = (unsigned)ga;                         // global addr [31:0]
  g0[3] = (unsigned)((ga >> 32) & 0x01FFFFFFu)  // global addr [56:32]
          | (2u << 30);                         // type = 2 ("image")
  v8u g1;
  g1[0] = (1u << 16)                            // data_size = 2 bytes
        | (1u << 20)                            // pad_enable
        | (pad_interval << 22) | (pad_amount << 25);
  g1[1] = (dim0 & 0xFFFFu) << 16;               // tensor_dim0[15:0]
  g1[2] = (dim0 >> 16) | ((dim1 & 0xFFFFu) << 16);   // dim0 hi | dim1 lo
  g1[3] = (dim1 >> 16) | ((dim0 & 0xFFFFu) << 16);   // dim1 hi | tile_dim0=dim0
  g1[4] = dim1 & 0xFFFFu;                       // tile_dim1 = dim1, tile_dim2 = 0
  g1[5] = stride0;                              // tensor_dim0_stride lo32
  g1[6] = 0u;
  g1[7] = 0u;
  asm volatile("tensor_load_to_lds %0, %1" :: "s"(g0), "s"(g1) : "memory");
}

// Load one 16x32 f16 WMMA operand fragment (ISA 7.12.2 layout);
// caller bakes the lane-group K-offset (0 or 8) into p. Two ds_load_b128.
__device__ __forceinline__ v16h load_frag(const _Float16* p) {
  union { v16h v; v8h h[2]; } u;
  u.h[0] = *(const v8h*)(p);
  u.h[1] = *(const v8h*)(p + 16);
  return u.v;
}

__device__ __forceinline__ v8f wmma_f16(v16h a, v16h b, v8f c) {
  return __builtin_amdgcn_wmma_f32_16x16x32_f16(false, a, false, b, (short)0, c,
                                                false, false);
}

__device__ __forceinline__ float silu(float x) { return x / (1.0f + __expf(-x)); }

// 16 x 16B async gather of one 256B f16 feature row into LDS
#define ASYNC_ROW(ldsoff, voff, base)                                          \
  do {                                                                         \
    _Pragma("clang diagnostic push")                                           \
    asm volatile("global_load_async_to_lds_b128 %0, %1, %2 offset:0"           \
                 :: "v"(ldsoff), "v"(voff), "s"(base) : "memory");             \
    asm volatile("global_load_async_to_lds_b128 %0, %1, %2 offset:16"          \
                 :: "v"(ldsoff), "v"(voff), "s"(base) : "memory");             \
    asm volatile("global_load_async_to_lds_b128 %0, %1, %2 offset:32"          \
                 :: "v"(ldsoff), "v"(voff), "s"(base) : "memory");             \
    asm volatile("global_load_async_to_lds_b128 %0, %1, %2 offset:48"          \
                 :: "v"(ldsoff), "v"(voff), "s"(base) : "memory");             \
    asm volatile("global_load_async_to_lds_b128 %0, %1, %2 offset:64"          \
                 :: "v"(ldsoff), "v"(voff), "s"(base) : "memory");             \
    asm volatile("global_load_async_to_lds_b128 %0, %1, %2 offset:80"          \
                 :: "v"(ldsoff), "v"(voff), "s"(base) : "memory");             \
    asm volatile("global_load_async_to_lds_b128 %0, %1, %2 offset:96"          \
                 :: "v"(ldsoff), "v"(voff), "s"(base) : "memory");             \
    asm volatile("global_load_async_to_lds_b128 %0, %1, %2 offset:112"         \
                 :: "v"(ldsoff), "v"(voff), "s"(base) : "memory");             \
    asm volatile("global_load_async_to_lds_b128 %0, %1, %2 offset:128"         \
                 :: "v"(ldsoff), "v"(voff), "s"(base) : "memory");             \
    asm volatile("global_load_async_to_lds_b128 %0, %1, %2 offset:144"         \
                 :: "v"(ldsoff), "v"(voff), "s"(base) : "memory");             \
    asm volatile("global_load_async_to_lds_b128 %0, %1, %2 offset:160"         \
                 :: "v"(ldsoff), "v"(voff), "s"(base) : "memory");             \
    asm volatile("global_load_async_to_lds_b128 %0, %1, %2 offset:176"         \
                 :: "v"(ldsoff), "v"(voff), "s"(base) : "memory");             \
    asm volatile("global_load_async_to_lds_b128 %0, %1, %2 offset:192"         \
                 :: "v"(ldsoff), "v"(voff), "s"(base) : "memory");             \
    asm volatile("global_load_async_to_lds_b128 %0, %1, %2 offset:208"         \
                 :: "v"(ldsoff), "v"(voff), "s"(base) : "memory");             \
    asm volatile("global_load_async_to_lds_b128 %0, %1, %2 offset:224"         \
                 :: "v"(ldsoff), "v"(voff), "s"(base) : "memory");             \
    asm volatile("global_load_async_to_lds_b128 %0, %1, %2 offset:240"         \
                 :: "v"(ldsoff), "v"(voff), "s"(base) : "memory");             \
    _Pragma("clang diagnostic pop")                                            \
  } while (0)

// ---------------- prep kernels ----------------

__global__ void convert_weights_kernel(const float* __restrict__ W1e,
                                       const float* __restrict__ W2e,
                                       const float* __restrict__ W1n,
                                       const float* __restrict__ W2n,
                                       _Float16* __restrict__ w1et,
                                       _Float16* __restrict__ w2et,
                                       _Float16* __restrict__ w1nt,
                                       _Float16* __restrict__ w2nt,
                                       float* __restrict__ wlast) {
  int tid = blockIdx.x * blockDim.x + threadIdx.x;
  if (tid < 32768) {                       // W1e[0:256,:] -> w1et[128][256]
    int n = tid >> 8, k = tid & 255;
    w1et[n * 256 + k] = (_Float16)W1e[k * 128 + n];
  } else if (tid < 49152) {                // W2e -> w2et[128][128]
    int t = tid - 32768; int n = t >> 7, k = t & 127;
    w2et[n * 128 + k] = (_Float16)W2e[k * 128 + n];
  } else if (tid < 81920) {                // W1n -> w1nt[128][256]
    int t = tid - 49152; int n = t >> 8, k = t & 255;
    w1nt[n * 256 + k] = (_Float16)W1n[k * 128 + n];
  } else if (tid < 98304) {                // W2n -> w2nt[128][128]
    int t = tid - 81920; int n = t >> 7, k = t & 127;
    w2nt[n * 128 + k] = (_Float16)W2n[k * 128 + n];
  } else if (tid < 98432) {                // dist2 row of W1e (kept f32)
    int n = tid - 98304;
    wlast[n] = W1e[256 * 128 + n];
  }
}

__global__ void cvt_f16x4_kernel(const float4* __restrict__ s,
                                 v4h* __restrict__ d, int n4) {
  int i = blockIdx.x * blockDim.x + threadIdx.x;
  if (i < n4) {
    float4 f = s[i];
    v4h h; h.x = (_Float16)f.x; h.y = (_Float16)f.y;
    h.z = (_Float16)f.z; h.w = (_Float16)f.w;
    d[i] = h;
  }
}

__global__ void zero_f4_kernel(float4* __restrict__ p, int n4) {
  int i = blockIdx.x * blockDim.x + threadIdx.x;
  if (i < n4) p[i] = make_float4(0.f, 0.f, 0.f, 0.f);
}

// LDS byte-offset layout constants (dynamic LDS starts at offset 0)
#define EDGE_WBYTES (128 * LDK * 2 + 128 * LDK2 * 2 + 3 * 128 * 4)  // 103936
#define EDGE_PWB    (16 * LDK * 2 + 16 * LDK2 * 2 + 16 * 4 + 16 * 4)
#define NODE_WBYTES (128 * LDK * 2 + 128 * LDK2 * 2 + 4 * 128 * 4)  // 104448
#define NODE_PWB    (16 * LDK * 2 + 16 * LDK2 * 2 + 16 * LDY * 4)

// ---------------- edge kernel: async gather -> MLP -> atomic scatter ----------------
__global__ __launch_bounds__(32 * WAVES) void egnn_edge_kernel(
    const _Float16* __restrict__ x16, const float* __restrict__ pos,
    const int* __restrict__ eidx,
    const float* __restrict__ b1e, const float* __restrict__ b2e,
    const _Float16* __restrict__ w1t, const _Float16* __restrict__ w2t,
    const float* __restrict__ wlast,
    float* __restrict__ agg, int nTiles) {
  extern __shared__ char smem[];
  _Float16* sW1 = (_Float16*)smem;              // [128][LDK]   @ 0
  _Float16* sW2 = sW1 + 128 * LDK;              // [128][LDK2]  @ 67584
  float* sWl = (float*)(sW2 + 128 * LDK2);
  float* sB1 = sWl + 128;
  float* sB2 = sB1 + 128;
  char* pw = (char*)(sB2 + 128);
  const int wave = threadIdx.x >> 5;
  const int lane = threadIdx.x & 31;
  _Float16* sA = (_Float16*)(pw + wave * EDGE_PWB);
  _Float16* sH = sA + 16 * LDK;
  float* sD2 = (float*)(sH + 16 * LDK2);
  int* sRow = (int*)(sD2 + 16);

  // ---- weight staging via Tensor Data Mover (one wave issues both tiles) ----
  if (wave == 0) {
    // W1^T: 128 rows x 256 halves, pad 4 dwords after every 128 dwords -> LDK
    tdm_load_2d(0u, w1t, 256u, 128u, 256u, /*interval=*/6u, /*amount=*/3u);
    // W2^T: 128 rows x 128 halves, pad 4 dwords after every 64 dwords -> LDK2
    tdm_load_2d((unsigned)(128 * LDK * 2), w2t, 128u, 128u, 128u, 5u, 3u);
    __builtin_amdgcn_s_wait_tensorcnt(0);
  }
  for (int i = threadIdx.x; i < 128; i += blockDim.x) {
    sWl[i] = wlast[i]; sB1[i] = b1e[i]; sB2[i] = b2e[i];
  }
  __syncthreads();

  const int m = lane & 15;             // row (A) / col (B,C) index
  const int kh = (lane >> 4) << 3;     // K-half select: 0 or 8
  const int mbase = (lane >> 4) << 3;  // C fragment row base: 0 or 8
  const int el = lane >> 1, which = lane & 1;
  const unsigned ldsA = (unsigned)(EDGE_WBYTES + wave * EDGE_PWB +
                                   el * (LDK * 2) + which * (D * 2));

  for (int t = blockIdx.x * WAVES + wave; t < nTiles; t += gridDim.x * WAVES) {
    // ---- stage 16 edges: f16 feature rows, memory -> LDS, no VGPR round-trip
    int ge = t * 16 + el;                     // E % 16 == 0: no guard
    int r = eidx[ge];
    int c = eidx[N_EDGES + ge];
    int node = which ? c : r;
    unsigned voff = (unsigned)node * (D * 2); // byte offset of f16 row
    ASYNC_ROW(ldsA, voff, x16);
    if (which == 0) {
      float dx = pos[r * 3 + 0] - pos[c * 3 + 0];
      float dy = pos[r * 3 + 1] - pos[c * 3 + 1];
      float dz = pos[r * 3 + 2] - pos[c * 3 + 2];
      sD2[el] = dx * dx + dy * dy + dz * dz;
      sRow[el] = r;
    }
    async_fence();
    lds_fence();

    float dvals[8]; int rows[8];
#pragma unroll
    for (int i = 0; i < 8; ++i) { dvals[i] = sD2[mbase + i]; rows[i] = sRow[mbase + i]; }

    // ---- layer 1: H = silu(A @ W1 + dist2 (x) wlast + b1),  K = 256 ----
    for (int tc = 0; tc < 8; ++tc) {
      v8f acc = {};
      const _Float16* ap = sA + m * LDK + kh;
      const _Float16* bp = sW1 + (tc * 16 + m) * LDK + kh;
#pragma unroll
      for (int k = 0; k < 8; ++k)
        acc = wmma_f16(load_frag(ap + k * 32), load_frag(bp + k * 32), acc);
      float bias = sB1[tc * 16 + m];
      float wl = sWl[tc * 16 + m];
      _Float16* hp = sH + tc * 16 + m;
#pragma unroll
      for (int i = 0; i < 8; ++i) {
        float v = silu(acc[i] + bias + dvals[i] * wl);
        hp[(mbase + i) * LDK2] = (_Float16)v;
      }
    }
    lds_fence();

    // ---- layer 2: E = silu(H @ W2 + b2), K = 128; fused scatter-add ----
    for (int tc = 0; tc < 8; ++tc) {
      v8f acc = {};
      const _Float16* ap = sH + m * LDK2 + kh;
      const _Float16* bp = sW2 + (tc * 16 + m) * LDK2 + kh;
#pragma unroll
      for (int k = 0; k < 4; ++k)
        acc = wmma_f16(load_frag(ap + k * 32), load_frag(bp + k * 32), acc);
      float bias = sB2[tc * 16 + m];
      int nc = tc * 16 + m;
#pragma unroll
      for (int i = 0; i < 8; ++i) {
        float v = silu(acc[i] + bias);
        __hip_atomic_fetch_add(&agg[(size_t)rows[i] * D + nc], v,
                               __ATOMIC_RELAXED, __HIP_MEMORY_SCOPE_AGENT);
      }
    }
    lds_fence();   // DS reads of sA/sH done before next tile's async stores
  }
}

// ---------------- node kernel: [x16|agg16] MLP -> residual -> LayerNorm ----------------
__global__ __launch_bounds__(32 * WAVES) void egnn_node_kernel(
    const float* __restrict__ x, const _Float16* __restrict__ x16,  // agg16 = x16 + N*D
    const float* __restrict__ b1n, const float* __restrict__ b2n,
    const float* __restrict__ gamma, const float* __restrict__ beta,
    const _Float16* __restrict__ w1t, const _Float16* __restrict__ w2t,
    float* __restrict__ out, int nTiles) {
  extern __shared__ char smem[];
  _Float16* sW1 = (_Float16*)smem;
  _Float16* sW2 = sW1 + 128 * LDK;
  float* sB1 = (float*)(sW2 + 128 * LDK2);
  float* sB2 = sB1 + 128;
  float* sG  = sB2 + 128;
  float* sBt = sG + 128;
  char* pw = (char*)(sBt + 128);
  const int wave = threadIdx.x >> 5;
  const int lane = threadIdx.x & 31;
  _Float16* sA = (_Float16*)(pw + wave * NODE_PWB);
  _Float16* sH = sA + 16 * LDK;
  float* sY = (float*)(sH + 16 * LDK2);

  if (wave == 0) {
    tdm_load_2d(0u, w1t, 256u, 128u, 256u, 6u, 3u);
    tdm_load_2d((unsigned)(128 * LDK * 2), w2t, 128u, 128u, 128u, 5u, 3u);
    __builtin_amdgcn_s_wait_tensorcnt(0);
  }
  for (int i = threadIdx.x; i < 128; i += blockDim.x) {
    sB1[i] = b1n[i]; sB2[i] = b2n[i]; sG[i] = gamma[i]; sBt[i] = beta[i];
  }
  __syncthreads();

  const int m = lane & 15;
  const int kh = (lane >> 4) << 3;
  const int mbase = (lane >> 4) << 3;
  const int el = lane >> 1, which = lane & 1;
  const unsigned ldsA = (unsigned)(NODE_WBYTES + wave * NODE_PWB +
                                   el * (LDK * 2) + which * (D * 2));
  const unsigned srcSel = (unsigned)which * (N_NODES * D * 2);  // x16 vs agg16

  for (int t = blockIdx.x * WAVES + wave; t < nTiles; t += gridDim.x * WAVES) {
    // ---- stage 16 nodes: [x16 | agg16] rows, memory -> LDS ----
    int gn = t * 16 + el;                     // N % 16 == 0
    unsigned voff = srcSel + (unsigned)gn * (D * 2);
    ASYNC_ROW(ldsA, voff, x16);
    async_fence();
    lds_fence();

    // ---- layer 1: H = silu([x|agg] @ W1n + b1n), K = 256 ----
    for (int tc = 0; tc < 8; ++tc) {
      v8f acc = {};
      const _Float16* ap = sA + m * LDK + kh;
      const _Float16* bp = sW1 + (tc * 16 + m) * LDK + kh;
#pragma unroll
      for (int k = 0; k < 8; ++k)
        acc = wmma_f16(load_frag(ap + k * 32), load_frag(bp + k * 32), acc);
      float bias = sB1[tc * 16 + m];
      _Float16* hp = sH + tc * 16 + m;
#pragma unroll
      for (int i = 0; i < 8; ++i) {
        float v = silu(acc[i] + bias);
        hp[(mbase + i) * LDK2] = (_Float16)v;
      }
    }
    lds_fence();

    // ---- layer 2 + residual: y = x + H @ W2n + b2n ----
    for (int tc = 0; tc < 8; ++tc) {
      v8f acc = {};
      const _Float16* ap = sH + m * LDK2 + kh;
      const _Float16* bp = sW2 + (tc * 16 + m) * LDK2 + kh;
#pragma unroll
      for (int k = 0; k < 4; ++k)
        acc = wmma_f16(load_frag(ap + k * 32), load_frag(bp + k * 32), acc);
      float bias = sB2[tc * 16 + m];
      int nc = tc * 16 + m;
#pragma unroll
      for (int i = 0; i < 8; ++i) {
        int mrow = mbase + i;
        float v = acc[i] + bias + x[(size_t)(t * 16 + mrow) * D + nc];
        sY[mrow * LDY + nc] = v;
      }
    }
    lds_fence();

    // ---- LayerNorm over D=128: 2 lanes per row, shuffle-combine ----
    {
      int mrow = lane & 15;
      int hsel = lane >> 4;
      const float* yrow = sY + mrow * LDY + hsel * 64;
      float s = 0.f, ss = 0.f;
#pragma unroll
      for (int j = 0; j < 64; ++j) { float v = yrow[j]; s += v; ss += v * v; }
      s  += __shfl_xor(s, 16);
      ss += __shfl_xor(ss, 16);
      float mu = s * (1.0f / 128.0f);
      float var = ss * (1.0f / 128.0f) - mu * mu;
      float rs = rsqrtf(var + EPS);
      float* orow = out + (size_t)(t * 16 + mrow) * D + hsel * 64;
#pragma unroll
      for (int j = 0; j < 64; ++j) {
        int cc = hsel * 64 + j;
        orow[j] = (yrow[j] - mu) * rs * sG[cc] + sBt[cc];
      }
    }
    lds_fence();
  }
}

// ---------------- host launcher ----------------
extern "C" void kernel_launch(void* const* d_in, const int* in_sizes, int n_in,
                              void* d_out, int out_size, void* d_ws, size_t ws_size,
                              hipStream_t stream) {
  const float* x    = (const float*)d_in[0];
  const float* pos  = (const float*)d_in[1];
  const int*   eidx = (const int*)d_in[2];
  const float* W1e  = (const float*)d_in[3];
  const float* b1e  = (const float*)d_in[4];
  const float* W2e  = (const float*)d_in[5];
  const float* b2e  = (const float*)d_in[6];
  const float* W1n  = (const float*)d_in[7];
  const float* b1n  = (const float*)d_in[8];
  const float* W2n  = (const float*)d_in[9];
  const float* b2n  = (const float*)d_in[10];
  const float* gam  = (const float*)d_in[11];
  const float* bet  = (const float*)d_in[12];
  float* out = (float*)d_out;

  // ws layout: agg f32 | x16 f16 | agg16 f16 (contiguous with x16) | weights f16 | wlast f32
  char* ws = (char*)d_ws;
  float* agg = (float*)ws;                                   // 25,600,000 B
  _Float16* x16   = (_Float16*)(ws + (size_t)N_NODES * D * 4);
  _Float16* agg16 = x16 + (size_t)N_NODES * D;               // contiguous after x16
  _Float16* w1et = agg16 + (size_t)N_NODES * D;
  _Float16* w2et = w1et + 128 * 256;
  _Float16* w1nt = w2et + 128 * 128;
  _Float16* w2nt = w1nt + 128 * 256;
  float* wlast = (float*)(w2nt + 128 * 128);

  convert_weights_kernel<<<(98432 + 255) / 256, 256, 0, stream>>>(
      W1e, W2e, W1n, W2n, w1et, w2et, w1nt, w2nt, wlast);

  int n4 = N_NODES * D / 4;  // 1,600,000 float4 groups
  cvt_f16x4_kernel<<<(n4 + 255) / 256, 256, 0, stream>>>(
      (const float4*)x, (v4h*)x16, n4);
  zero_f4_kernel<<<(n4 + 255) / 256, 256, 0, stream>>>((float4*)agg, n4);

  const size_t ldsE = (size_t)EDGE_WBYTES + (size_t)WAVES * EDGE_PWB;
  const int edgeTiles = N_EDGES / 16;  // 50000
  egnn_edge_kernel<<<1024, 32 * WAVES, ldsE, stream>>>(
      x16, pos, eidx, b1e, b2e, w1et, w2et, wlast, agg, edgeTiles);

  cvt_f16x4_kernel<<<(n4 + 255) / 256, 256, 0, stream>>>(
      (const float4*)agg, (v4h*)agg16, n4);

  const size_t ldsN = (size_t)NODE_WBYTES + (size_t)WAVES * NODE_PWB;
  const int nodeTiles = N_NODES / 16;  // 3125
  int nodeBlocks = (nodeTiles + WAVES - 1) / WAVES;
  egnn_node_kernel<<<nodeBlocks, 32 * WAVES, ldsN, stream>>>(
      x, x16, b1n, b2n, gam, bet, w1nt, w2nt, out, nodeTiles);
}
